// CausalSelfAttention_89842125897794
// MI455X (gfx1250) — matrix operand order repaired and checked by reading, not compiled, not run on hardware
//
#include <hip/hip_runtime.h>
#include <hip/hip_bf16.h>
#include <stdint.h>

#define B_ 2
#define T_ 2048
#define C_ 1024
#define H_ 16
#define D_ 64
#define M_TOT (B_ * T_)   // 4096 tokens
#define N_QKV (3 * C_)    // 3072

typedef __bf16 bf16_t;
typedef bf16_t bf16x16 __attribute__((ext_vector_type(16)));
typedef float  f32x8   __attribute__((ext_vector_type(8)));

union FragB16 {
    bf16x16  v;
    uint4    q2[2];
    uint32_t u[8];
};
union FragF32 {
    f32x8 v;
    float f[8];
};

// A-matrix fragment (16xK=32, 16-bit): lane<16 holds K 0..15, lane>=16 holds K 16..31 contiguous.
// Source row-major, element (row, k) contiguous in k.
__device__ __forceinline__ void load_a_frag(FragB16& f, const bf16_t* base, int row,
                                            int stride, int k0, int lane) {
    const bf16_t* p = base + (size_t)row * stride + k0 + ((lane >> 4) << 4);
    f.q2[0] = *(const uint4*)p;
    f.q2[1] = *(const uint4*)(p + 8);
}

// B-matrix fragment (K=32 x 16, 16-bit): lane n=lane%16 holds K = 16*(lane/16) + 0..15 contiguous.
// Source "row-major by n": element (n, k) contiguous in k (i.e. B^T row-major).
__device__ __forceinline__ void load_b_frag(FragB16& f, const bf16_t* base, int nrow,
                                            int stride, int k0, int lane) {
    const bf16_t* p = base + (size_t)nrow * stride + k0 + ((lane >> 4) << 4);
    f.q2[0] = *(const uint4*)p;
    f.q2[1] = *(const uint4*)(p + 8);
}

__device__ __forceinline__ f32x8 wmma_bf16(const FragB16& a, const FragB16& b, f32x8 c) {
    return __builtin_amdgcn_wmma_f32_16x16x32_bf16(false, a.v, false, b.v,
                                                   (short)0, c, false, false);
}

__device__ __forceinline__ uint32_t pack_bf16(float a, float b) {
    bf16_t x = (bf16_t)a, y = (bf16_t)b;
    uint16_t ux = __builtin_bit_cast(uint16_t, x);
    uint16_t uy = __builtin_bit_cast(uint16_t, y);
    return (uint32_t)ux | ((uint32_t)uy << 16);
}

// ---------------- conversion kernels ----------------
__global__ void cvt_f32_bf16_k(const float* __restrict__ in, bf16_t* __restrict__ out, int n) {
    int i = blockIdx.x * 256 + threadIdx.x;
    if (i < n) out[i] = (bf16_t)in[i];
}

// in [K][N] fp32 -> out [N][K] bf16
__global__ void transpose_cvt_k(const float* __restrict__ in, bf16_t* __restrict__ out,
                                int K, int N) {
    int i = blockIdx.x * 256 + threadIdx.x;
    if (i < K * N) {
        int n = i / K;
        int k = i - n * K;
        out[i] = (bf16_t)in[(size_t)k * N + n];
    }
}

// ---------------- QKV projection GEMM (32x64 tile per wave) ----------------
// xb [M_TOT][C_] bf16, wT [N_QKV][C_] bf16 (w_qkv transposed), bias fp32 [N_QKV]
// Writes q (pre-scaled by D^-0.5), k as [b,h,t,d] bf16 and v as [b,h,d,t] bf16.
__global__ void qkv_gemm_k(const bf16_t* __restrict__ xb, const bf16_t* __restrict__ wT,
                           const float* __restrict__ bias,
                           bf16_t* __restrict__ qA, bf16_t* __restrict__ kA,
                           bf16_t* __restrict__ vA) {
    const int lane = threadIdx.x;
    const int m0 = blockIdx.x << 5;
    const int n0 = blockIdx.y << 6;

    FragF32 acc[2][4];
#pragma unroll
    for (int mi = 0; mi < 2; ++mi)
#pragma unroll
        for (int s = 0; s < 4; ++s)
#pragma unroll
            for (int r = 0; r < 8; ++r) acc[mi][s].f[r] = 0.f;

    for (int kk = 0; kk < C_; kk += 32) {
        FragB16 a0, a1;
        load_a_frag(a0, xb, m0 + (lane & 15), C_, kk, lane);
        load_a_frag(a1, xb, m0 + 16 + (lane & 15), C_, kk, lane);
#pragma unroll
        for (int s = 0; s < 4; ++s) {
            FragB16 b;
            load_b_frag(b, wT, n0 + (s << 4) + (lane & 15), C_, kk, lane);
            acc[0][s].v = wmma_bf16(a0, b, acc[0][s].v);
            acc[1][s].v = wmma_bf16(a1, b, acc[1][s].v);
        }
    }

    const int which = n0 >> 10;            // 0=q 1=k 2=v
    const int h = (n0 & 1023) >> 6;
#pragma unroll
    for (int s = 0; s < 4; ++s) {
        const int nglob = n0 + (s << 4) + (lane & 15);
        const int d = (s << 4) + (lane & 15);
        const float bv = bias[nglob];
#pragma unroll
        for (int mi = 0; mi < 2; ++mi)
#pragma unroll
            for (int r = 0; r < 8; ++r) {
                const int m = m0 + (mi << 4) + r + ((lane >> 4) << 3);
                const int b = m >> 11;          // /T_
                const int t = m & (T_ - 1);
                const float val = acc[mi][s].f[r] + bv;
                const size_t bh = (size_t)(b * H_ + h);
                if (which == 0) {
                    qA[(bh * T_ + t) * D_ + d] = (bf16_t)(val * 0.125f);  // D^-0.5
                } else if (which == 1) {
                    kA[(bh * T_ + t) * D_ + d] = (bf16_t)val;
                } else {
                    vA[(bh * D_ + d) * T_ + t] = (bf16_t)val;             // V transposed
                }
            }
    }
}

// ---------------- flash attention (one wave per (b,h,32 q rows)) ----------------
__global__ void flash_attn_k(const bf16_t* __restrict__ qArr, const bf16_t* __restrict__ kArr,
                             const bf16_t* __restrict__ vArr, bf16_t* __restrict__ attnOut) {
    const int lane = threadIdx.x;
    const int qt = blockIdx.x & 63;         // T_/32 = 64
    const int bh = blockIdx.x >> 6;
    const int qBase = qt << 5;

    const bf16_t* Q = qArr + (size_t)bh * T_ * D_;
    const bf16_t* K = kArr + (size_t)bh * T_ * D_;
    const bf16_t* V = vArr + (size_t)bh * D_ * T_;   // [d][t]

    // Q as B-matrix (Q^T, d x q): two q sub-tiles x two 32-d chunks; reused for all k.
    FragB16 bq[2][2];
#pragma unroll
    for (int qs = 0; qs < 2; ++qs)
#pragma unroll
        for (int dc = 0; dc < 2; ++dc)
            load_b_frag(bq[qs][dc], Q, qBase + (qs << 4) + (lane & 15), D_, dc * 32, lane);

    FragF32 o[2][4];
#pragma unroll
    for (int qs = 0; qs < 2; ++qs)
#pragma unroll
        for (int t = 0; t < 4; ++t)
#pragma unroll
            for (int r = 0; r < 8; ++r) o[qs][t].f[r] = 0.f;
    float mrow[2] = {-INFINITY, -INFINITY};
    float lrow[2] = {0.f, 0.f};

    const int qgl = lane & 15;
    const int klo = (lane >> 4) << 3;
    const int nkb = qt + 1;                 // 32-wide k blocks intersecting causal range

    for (int kb = 0; kb < nkb; ++kb) {
        const int k0 = kb << 5;

        // prefetch next 32-k block of K (32 rows) and V (64 row segments) into cache
        __builtin_prefetch(K + (size_t)(k0 + 32 + lane) * D_, 0, 3);
        __builtin_prefetch(V + (size_t)lane * T_ + k0 + 32, 0, 3);
        __builtin_prefetch(V + (size_t)(lane + 32) * T_ + k0 + 32, 0, 3);

        FragF32 s0[2], s1[2];
#pragma unroll
        for (int qs = 0; qs < 2; ++qs)
#pragma unroll
            for (int r = 0; r < 8; ++r) { s0[qs].f[r] = 0.f; s1[qs].f[r] = 0.f; }

        // S^T tiles = K_tile(16k x 64d) x Q^T(64d x 16q); K frags shared by both q sub-tiles
#pragma unroll
        for (int dc = 0; dc < 2; ++dc) {
            FragB16 ak0, ak1;
            load_a_frag(ak0, K, k0 + (lane & 15), D_, dc * 32, lane);
            load_a_frag(ak1, K, k0 + 16 + (lane & 15), D_, dc * 32, lane);
#pragma unroll
            for (int qs = 0; qs < 2; ++qs) {
                s0[qs].v = wmma_bf16(ak0, bq[qs][dc], s0[qs].v);
                s1[qs].v = wmma_bf16(ak1, bq[qs][dc], s1[qs].v);
            }
        }

        FragB16 pb[2];
#pragma unroll
        for (int qs = 0; qs < 2; ++qs) {
            const int qg = qBase + (qs << 4) + qgl;
            // causal mask: element (k = k0 + klo + r [+16], q = qg)
#pragma unroll
            for (int r = 0; r < 8; ++r) {
                s0[qs].f[r] = (k0 + klo + r > qg) ? -INFINITY : s0[qs].f[r];
                s1[qs].f[r] = (k0 + 16 + klo + r > qg) ? -INFINITY : s1[qs].f[r];
            }
            // online softmax (row q lives on lane and lane^16)
            float tmax = -INFINITY;
#pragma unroll
            for (int r = 0; r < 8; ++r)
                tmax = fmaxf(tmax, fmaxf(s0[qs].f[r], s1[qs].f[r]));
            tmax = fmaxf(tmax, __shfl_xor(tmax, 16, 32));
            const float mnew = fmaxf(mrow[qs], tmax);
            const float alpha = __expf(mrow[qs] - mnew);
            float psum = 0.f;
#pragma unroll
            for (int r = 0; r < 8; ++r) {
                s0[qs].f[r] = __expf(s0[qs].f[r] - mnew); psum += s0[qs].f[r];
                s1[qs].f[r] = __expf(s1[qs].f[r] - mnew); psum += s1[qs].f[r];
            }
            psum += __shfl_xor(psum, 16, 32);
            lrow[qs] = lrow[qs] * alpha + psum;
            mrow[qs] = mnew;
#pragma unroll
            for (int t = 0; t < 4; ++t)
#pragma unroll
                for (int r = 0; r < 8; ++r) o[qs][t].f[r] *= alpha;

            // Build P^T B-fragment: lane must hold k = 16*(lane/16)+0..15; C layout
            // interleaves halves by 8, so swap the "other 8" with the partner lane.
            uint32_t pk0[4], pk1[4];
#pragma unroll
            for (int j = 0; j < 4; ++j) {
                pk0[j] = pack_bf16(s0[qs].f[2 * j], s0[qs].f[2 * j + 1]);
                pk1[j] = pack_bf16(s1[qs].f[2 * j], s1[qs].f[2 * j + 1]);
            }
            const bool hi = (lane >= 16);
#pragma unroll
            for (int j = 0; j < 4; ++j) {
                const uint32_t x0 = __shfl_xor(pk0[j], 16, 32);
                const uint32_t x1 = __shfl_xor(pk1[j], 16, 32);
                pb[qs].u[j]     = hi ? x1 : pk0[j];   // lane<16: k0..7  ; lane>=16: k16..23
                pb[qs].u[4 + j] = hi ? pk1[j] : x0;   // lane<16: k8..15 ; lane>=16: k24..31
            }
        }

        // O^T += V^T_tile(16d x 32k) x P^T(32k x 16q); V frags shared by both q sub-tiles
#pragma unroll
        for (int dt = 0; dt < 4; ++dt) {
            FragB16 av;
            load_a_frag(av, V, (dt << 4) + (lane & 15), T_, k0, lane);
#pragma unroll
            for (int qs = 0; qs < 2; ++qs)
                o[qs][dt].v = wmma_bf16(av, pb[qs], o[qs][dt].v);
        }
    }

    // epilogue: O[q][d] = O^T / l, written to [b*T+t][h*64+d] bf16
    const int b = bh >> 4, h = bh & 15;
#pragma unroll
    for (int qs = 0; qs < 2; ++qs) {
        const float linv = 1.0f / lrow[qs];
        const int qg = qBase + (qs << 4) + qgl;
        const size_t rowbase = ((size_t)(b * T_ + qg)) * C_ + h * D_;
#pragma unroll
        for (int dt = 0; dt < 4; ++dt)
#pragma unroll
            for (int j = 0; j < 4; ++j) {
                const float v0 = o[qs][dt].f[2 * j] * linv;
                const float v1 = o[qs][dt].f[2 * j + 1] * linv;
                const int d = (dt << 4) + klo + 2 * j;
                *(uint32_t*)(attnOut + rowbase + d) = pack_bf16(v0, v1);
            }
    }
}

// ---------------- output projection GEMM (32x64 tile per wave) ----------------
__global__ void out_gemm_k(const bf16_t* __restrict__ attnb, const bf16_t* __restrict__ wT,
                           const float* __restrict__ bias, float* __restrict__ out) {
    const int lane = threadIdx.x;
    const int m0 = blockIdx.x << 5;
    const int n0 = blockIdx.y << 6;

    FragF32 acc[2][4];
#pragma unroll
    for (int mi = 0; mi < 2; ++mi)
#pragma unroll
        for (int s = 0; s < 4; ++s)
#pragma unroll
            for (int r = 0; r < 8; ++r) acc[mi][s].f[r] = 0.f;

    for (int kk = 0; kk < C_; kk += 32) {
        FragB16 a0, a1;
        load_a_frag(a0, attnb, m0 + (lane & 15), C_, kk, lane);
        load_a_frag(a1, attnb, m0 + 16 + (lane & 15), C_, kk, lane);
#pragma unroll
        for (int s = 0; s < 4; ++s) {
            FragB16 b;
            load_b_frag(b, wT, n0 + (s << 4) + (lane & 15), C_, kk, lane);
            acc[0][s].v = wmma_bf16(a0, b, acc[0][s].v);
            acc[1][s].v = wmma_bf16(a1, b, acc[1][s].v);
        }
    }

#pragma unroll
    for (int s = 0; s < 4; ++s) {
        const int n = n0 + (s << 4) + (lane & 15);
        const float bv = bias[n];
#pragma unroll
        for (int mi = 0; mi < 2; ++mi)
#pragma unroll
            for (int r = 0; r < 8; ++r) {
                const int m = m0 + (mi << 4) + r + ((lane >> 4) << 3);
                out[(size_t)m * C_ + n] = acc[mi][s].f[r] + bv;
            }
    }
}

extern "C" void kernel_launch(void* const* d_in, const int* in_sizes, int n_in,
                              void* d_out, int out_size, void* d_ws, size_t ws_size,
                              hipStream_t stream) {
    const float* x     = (const float*)d_in[0];
    const float* w_qkv = (const float*)d_in[1];
    const float* b_qkv = (const float*)d_in[2];
    const float* w_out = (const float*)d_in[3];
    const float* b_out = (const float*)d_in[4];
    float* out = (float*)d_out;

    bf16_t* xb    = (bf16_t*)d_ws;                        // 4M elems
    bf16_t* wqkvT = xb    + (size_t)M_TOT * C_;           // 3M
    bf16_t* woutT = wqkvT + (size_t)N_QKV * C_;           // 1M
    bf16_t* qA    = woutT + (size_t)C_ * C_;              // 4M
    bf16_t* kA    = qA    + (size_t)B_ * H_ * T_ * D_;    // 4M
    bf16_t* vA    = kA    + (size_t)B_ * H_ * T_ * D_;    // 4M
    bf16_t* attnb = vA    + (size_t)B_ * H_ * T_ * D_;    // 4M   (48MB total)

    cvt_f32_bf16_k<<<(M_TOT * C_) / 256, 256, 0, stream>>>(x, xb, M_TOT * C_);
    transpose_cvt_k<<<(N_QKV * C_) / 256, 256, 0, stream>>>(w_qkv, wqkvT, C_, N_QKV);
    transpose_cvt_k<<<(C_ * C_) / 256, 256, 0, stream>>>(w_out, woutT, C_, C_);

    qkv_gemm_k<<<dim3(M_TOT / 32, N_QKV / 64), 32, 0, stream>>>(xb, wqkvT, b_qkv, qA, kA, vA);
    flash_attn_k<<<B_ * H_ * (T_ / 32), 32, 0, stream>>>(qA, kA, vA, attnb);
    out_gemm_k<<<dim3(M_TOT / 32, C_ / 64), 32, 0, stream>>>(attnb, woutT, b_out, out);
}
